// GeometricCrossAttention_59485297050097
// MI455X (gfx1250) — matrix operand order repaired and checked by reading, not compiled, not run on hardware
//
#include <hip/hip_runtime.h>
#include <hip/hip_bf16.h>

typedef __attribute__((ext_vector_type(16))) _Float16 v16h;
typedef __attribute__((ext_vector_type(8)))  _Float16 v8h;
typedef __attribute__((ext_vector_type(8)))  float    v8f;

#define NN   1024
#define HID  256
#define NH   8
#define HD   32
#define NE   16384

__device__ inline v8f wmma_f16(v16h a, v16h b, v8f c) {
    return __builtin_amdgcn_wmma_f32_16x16x32_f16(
        /*neg_a=*/false, a, /*neg_b=*/false, b,
        /*c_mod=*/(short)0, c, /*reuse_a=*/false, /*reuse_b=*/false);
}

__device__ inline v16h cat2(v8h lo, v8h hi) {
    return __builtin_shufflevector(lo, hi, 0, 1, 2, 3, 4, 5, 6, 7,
                                   8, 9, 10, 11, 12, 13, 14, 15);
}

// ---- WMMA operand loaders (f16 memory, contiguous k = 16*(lane>>4) + s) ------------------
// A 16x32 (MxK): lane holds row (lane&15); halves [koff..koff+15],
// koff = 16*(lane>=16). Two contiguous 16B loads.
__device__ inline v16h load_a_h(const _Float16* __restrict__ A, int lda, int row0, int k0, int lane) {
    const _Float16* p = A + (size_t)(row0 + (lane & 15)) * lda + k0 + ((lane >> 4) << 4);
    return cat2(*(const v8h*)p, *(const v8h*)(p + 8));
}

// B tile for K^T in attention: logical B[d][m] = kH[m*HID + hbase + d];
// per lane 16 contiguous halves.
__device__ inline v16h load_kT_bh(const _Float16* __restrict__ km, int col0, int hbase, int lane) {
    const _Float16* p = km + (size_t)(col0 + (lane & 15)) * HID + hbase + ((lane >> 4) << 4);
    return cat2(*(const v8h*)p, *(const v8h*)(p + 8));
}

// B tile for V from transposed layout vT[(h*32+d)][m]: per lane 16 contiguous
// halves along m.
__device__ inline v16h load_vT_bh(const _Float16* __restrict__ vt, int m0, int hbase, int d0, int lane) {
    const _Float16* p = vt + (size_t)(hbase + d0 + (lane & 15)) * NN + m0 + ((lane >> 4) << 4);
    return cat2(*(const v8h*)p, *(const v8h*)(p + 8));
}

// ---- Tiled f16-WMMA GEMM ----------------------------------------------------
// Block = 128x64 tile (8 waves x 16 rows, 4 accumulators/wave).
// A: f16 row-major [M][K].  BT: f16 pre-transposed [N][K] -> the k-slab is a
// byte-for-byte copy into LDS, staged with GLOBAL_LOAD_ASYNC_TO_LDS_B128
// (one b128 per thread), double-buffered so the DMA of slab k+32 overlaps
// the WMMAs on slab k; drained with s_wait_asynccnt.
// EPI: 0 = +bias, 1 = silu(+bias), 2 = min(aux, A@B).
// WF: write f32 C.  WMODE: 0 none, 1 f16 mirror CH[row][col],
//                          2 transposed f16 mirror CH[col][row].
template <int EPI, bool WF, int WMODE>
__global__ void gemm_kernel(const _Float16* __restrict__ A, const _Float16* __restrict__ BT,
                            const float* __restrict__ bias, const float* __restrict__ aux,
                            float* __restrict__ C, _Float16* __restrict__ CH,
                            int M, int N, int K) {
    __shared__ _Float16 sBt[2][64][40];   // double-buffered [col][k] slab (+pad)
    int lane = threadIdx.x & 31;
    int wave = threadIdx.x >> 5;
    int nbx = N >> 6;
    int bm = blockIdx.x / nbx;
    int bn = blockIdx.x - bm * nbx;
    int row0 = (bm << 7) + (wave << 4);
    int col0 = bn << 6;

    int t = threadIdx.x;
    int scol = t >> 2, sch = t & 3;       // 64 cols x 4 chunks of 8 halves
    const _Float16* gsrc = BT + (size_t)(col0 + scol) * K + sch * 8;
    unsigned ldso = (unsigned)(unsigned long long)(const void*)&sBt[0][scol][sch * 8];
    const unsigned bufstride = (unsigned)(64 * 40 * sizeof(_Float16));

    int cl = lane & 15;
    int koff = (lane >> 4) << 4;
    v8f acc[4] = {{}, {}, {}, {}};

    // prologue: stage slab 0
    asm volatile("global_load_async_to_lds_b128 %0, %1, off"
                 :: "v"(ldso), "v"(gsrc) : "memory");
    asm volatile("s_wait_asynccnt 0x0" ::: "memory");
    __syncthreads();

    for (int k0 = 0; k0 < K; k0 += 32) {
        int cb = (k0 >> 5) & 1;
        if (k0 + 32 < K) {   // prefetch next slab into alternate buffer
            asm volatile("global_load_async_to_lds_b128 %0, %1, off"
                         :: "v"(ldso + (unsigned)(cb ^ 1) * bufstride),
                            "v"(gsrc + k0 + 32) : "memory");
        }
        v16h a = load_a_h(A, K, row0, k0, lane);
#pragma unroll
        for (int j = 0; j < 4; ++j) {
            const _Float16* p = &sBt[cb][(j << 4) + cl][koff];
            v16h b = cat2(*(const v8h*)p, *(const v8h*)(p + 8));
            acc[j] = wmma_f16(a, b, acc[j]);
        }
        asm volatile("s_wait_asynccnt 0x0" ::: "memory");
        __syncthreads();
    }

    // C layout: VGPR r -> row row0 + r + 8*(lane>=16), col col0 + 16j + (lane&15)
    int rb = row0 + ((lane >> 4) << 3);
#pragma unroll
    for (int j = 0; j < 4; ++j) {
        int cc = col0 + (j << 4) + cl;
        float bv = (EPI == 2) ? 0.f : bias[cc];
#pragma unroll
        for (int r = 0; r < 8; ++r) {
            float v = acc[j][r] + bv;
            if (EPI == 1) v = v / (1.f + __expf(-v));
            if (EPI == 2) v = fminf(aux[(size_t)(rb + r) * N + cc], acc[j][r]);
            if (WF) C[(size_t)(rb + r) * N + cc] = v;
            if (WMODE == 1) CH[(size_t)(rb + r) * N + cc] = (_Float16)v;
            if (WMODE == 2) CH[(size_t)cc * M + (rb + r)] = (_Float16)v;
        }
    }
}

// ---- conversion / utility kernels ------------------------------------------
__global__ void zero_kernel(float* __restrict__ p, int n) {
    for (int i = blockIdx.x * blockDim.x + threadIdx.x; i < n; i += gridDim.x * blockDim.x)
        p[i] = 0.f;
}

__global__ void scatter_kernel(const int* __restrict__ ei, float* __restrict__ adj) {
    int e = blockIdx.x * 256 + threadIdx.x;
    if (e < NE) adj[(size_t)ei[e] * NN + ei[NE + e]] = 1.0f;
}

__global__ void cvt_h_kernel(const float* __restrict__ in, _Float16* __restrict__ out, int n) {
    int i = blockIdx.x * 256 + threadIdx.x;
    if (i < n) out[i] = (_Float16)in[i];
}

// out[n*K+k] = (f16) in[k*N+n]   (pre-transposed B operand)
__global__ void transpose_h_kernel(const float* __restrict__ in, _Float16* __restrict__ out,
                                   int K, int N) {
    int t = blockIdx.x * 256 + threadIdx.x;
    int n = t / K, k = t - n * K;
    out[t] = (_Float16)in[(size_t)k * N + n];
}

__global__ void cat_h_kernel(const float* __restrict__ a, const float* __restrict__ b,
                             _Float16* __restrict__ o) {
    int t = blockIdx.x * 256 + threadIdx.x;   // 1024*512
    int n = t >> 9, c = t & 511;
    o[t] = (_Float16)((c < HID) ? a[n * HID + c] : b[n * HID + c - HID]);
}

// gate logits + 2-class softmax
__global__ void gate2_kernel(const float* __restrict__ gh, const float* __restrict__ w2,
                             const float* __restrict__ b2, float* __restrict__ gate) {
    int n = blockIdx.x * 256 + threadIdx.x;
    const float* p = gh + (size_t)n * 1024;
    float z0 = b2[0], z1 = b2[1];
    for (int j = 0; j < 1024; ++j) {
        float h = p[j];
        z0 += h * w2[2 * j];
        z1 += h * w2[2 * j + 1];
    }
    float mx = fmaxf(z0, z1);
    float e0 = __expf(z0 - mx), e1 = __expf(z1 - mx);
    float inv = 1.f / (e0 + e1);
    gate[n * 2] = e0 * inv;
    gate[n * 2 + 1] = e1 * inv;
}

// per-pair geo + topo MLPs, gate-mixed -> bias[n][h][m]
__global__ void biasfuse_kernel(const float* __restrict__ pos, const float* __restrict__ path,
                                const float* __restrict__ gate,
                                const float* __restrict__ gw1, const float* __restrict__ gb1,
                                const float* __restrict__ gw2, const float* __restrict__ gb2,
                                const float* __restrict__ tw1, const float* __restrict__ tb1,
                                const float* __restrict__ tw2, const float* __restrict__ tb2,
                                float* __restrict__ outb) {
    __shared__ float s_gw1[32], s_gb1[32], s_gw2[256], s_gb2[8];
    __shared__ float s_tw1[32], s_tb1[32], s_tw2[256], s_tb2[8];
    int t = threadIdx.x;
    if (t < 32) { s_gw1[t] = gw1[t]; s_gb1[t] = gb1[t]; s_tw1[t] = tw1[t]; s_tb1[t] = tb1[t]; }
    if (t < 8)  { s_gb2[t] = gb2[t]; s_tb2[t] = tb2[t]; }
    if (t < 256) { s_gw2[t] = gw2[t]; s_tw2[t] = tw2[t]; }
    __syncthreads();

    int idx = blockIdx.x * 256 + t;
    int n = idx >> 10, m = idx & 1023;
    float dx = pos[n * 3]     - pos[m * 3];
    float dy = pos[n * 3 + 1] - pos[m * 3 + 1];
    float dz = pos[n * 3 + 2] - pos[m * 3 + 2];
    float dist = sqrtf(fmaxf(dx * dx + dy * dy + dz * dz, 1e-12f));
    float pv = path[(size_t)n * NN + m];

    float gb[8], tb[8];
#pragma unroll
    for (int h = 0; h < 8; ++h) { gb[h] = s_gb2[h]; tb[h] = s_tb2[h]; }
    for (int j = 0; j < 32; ++j) {
        float hg = dist * s_gw1[j] + s_gb1[j]; hg = hg / (1.f + __expf(-hg));
        float ht = pv   * s_tw1[j] + s_tb1[j]; ht = ht / (1.f + __expf(-ht));
#pragma unroll
        for (int h = 0; h < 8; ++h) {
            gb[h] += hg * s_gw2[j * 8 + h];
            tb[h] += ht * s_tw2[j * 8 + h];
        }
    }
    float g0 = gate[n * 2], g1 = gate[n * 2 + 1];
#pragma unroll
    for (int h = 0; h < 8; ++h)
        outb[((size_t)n * 8 + h) * NN + m] = g0 * gb[h] + g1 * tb[h];
}

// ---- fused flash attention: block = 16 query rows, wave = head --------------
__global__ void attn_kernel(const _Float16* __restrict__ q, const _Float16* __restrict__ kmat,
                            const _Float16* __restrict__ vt, const float* __restrict__ bias,
                            _Float16* __restrict__ outh) {
    __shared__ _Float16 sP[8][16][40];   // per-wave 16x32 prob tile (+pad)
    int lane = threadIdx.x & 31;
    int w = threadIdx.x >> 5;            // wave id == head
    int h = w;
    int n0 = blockIdx.x << 4;
    int c = lane & 15;
    int hi = lane >> 4;
    const float scale = 0.17677669529663687f;  // 1/sqrt(32)

    v16h qa = load_a_h(q, HID, n0, h * HD, lane);   // 16x32 A tile, K = D = 32
    v8f o0 = {}, o1 = {};
    float mrow[8], lrow[8];
#pragma unroll
    for (int j = 0; j < 8; ++j) { mrow[j] = -1e30f; lrow[j] = 0.f; }

    for (int m0 = 0; m0 < NN; m0 += 32) {
        v8f z = {};
        v8f s0 = wmma_f16(qa, load_kT_bh(kmat, m0,      h * HD, lane), z);
        v8f s1 = wmma_f16(qa, load_kT_bh(kmat, m0 + 16, h * HD, lane), z);

        float e0[8], e1[8];
#pragma unroll
        for (int j = 0; j < 8; ++j) {
            int r = n0 + j + (hi << 3);
            const float* bp = bias + ((size_t)r * 8 + h) * NN + m0 + c;
            s0[j] = s0[j] * scale + bp[0];
            s1[j] = s1[j] * scale + bp[16];
            float mx = fmaxf(s0[j], s1[j]);
#pragma unroll
            for (int d = 1; d < 16; d <<= 1) mx = fmaxf(mx, __shfl_xor(mx, d, 32));
            float mn = fmaxf(mrow[j], mx);
            float alpha = __expf(mrow[j] - mn);
            e0[j] = __expf(s0[j] - mn);
            e1[j] = __expf(s1[j] - mn);
            float rs = e0[j] + e1[j];
#pragma unroll
            for (int d = 1; d < 16; d <<= 1) rs += __shfl_xor(rs, d, 32);
            lrow[j] = lrow[j] * alpha + rs;
            o0[j] *= alpha;
            o1[j] *= alpha;
            mrow[j] = mn;
            int rr = j + (hi << 3);
            sP[w][rr][c]      = (_Float16)e0[j];   // C-layout -> LDS
            sP[w][rr][16 + c] = (_Float16)e1[j];
        }
        // read back in A layout (16x32)
        v16h pa;
        int r = lane & 15;
        int ko = hi << 4;
#pragma unroll
        for (int i = 0; i < 8; ++i) {
            pa[i]     = sP[w][r][ko + i];
            pa[8 + i] = sP[w][r][ko + 8 + i];
        }
        o0 = wmma_f16(pa, load_vT_bh(vt, m0, h * HD, 0,  lane), o0);
        o1 = wmma_f16(pa, load_vT_bh(vt, m0, h * HD, 16, lane), o1);
    }
#pragma unroll
    for (int j = 0; j < 8; ++j) {
        int r = n0 + j + (hi << 3);
        float inv = 1.f / lrow[j];
        outh[(size_t)r * HID + h * HD + c]      = (_Float16)(o0[j] * inv);
        outh[(size_t)r * HID + h * HD + 16 + c] = (_Float16)(o1[j] * inv);
    }
}

// ---- residual + LayerNorm ---------------------------------------------------
__global__ void resid_ln_kernel(const float* __restrict__ src, const float* __restrict__ proj,
                                const float* __restrict__ g, const float* __restrict__ b,
                                float* __restrict__ out) {
    int lane = threadIdx.x & 31;
    int w = threadIdx.x >> 5;
    int row = blockIdx.x * 8 + w;
    float x[8], s = 0.f, s2 = 0.f;
#pragma unroll
    for (int i = 0; i < 8; ++i) {
        int c = lane + 32 * i;
        x[i] = src[(size_t)row * HID + c] + proj[(size_t)row * HID + c];
        s += x[i];
        s2 += x[i] * x[i];
    }
#pragma unroll
    for (int d = 1; d < 32; d <<= 1) { s += __shfl_xor(s, d, 32); s2 += __shfl_xor(s2, d, 32); }
    float mu = s * (1.f / 256.f);
    float var = s2 * (1.f / 256.f) - mu * mu;
    float inv = rsqrtf(var + 1e-5f);
#pragma unroll
    for (int i = 0; i < 8; ++i) {
        int c = lane + 32 * i;
        out[(size_t)row * HID + c] = (x[i] - mu) * inv * g[c] + b[c];
    }
}

extern "C" void kernel_launch(void* const* d_in, const int* in_sizes, int n_in,
                              void* d_out, int out_size, void* d_ws, size_t ws_size,
                              hipStream_t stream) {
    const float* src  = (const float*)d_in[0];
    const float* tgt  = (const float*)d_in[1];
    const float* pos  = (const float*)d_in[2];
    const int*   ei   = (const int*)d_in[3];
    const float* Wq   = (const float*)d_in[4];  const float* bq = (const float*)d_in[5];
    const float* Wk   = (const float*)d_in[6];  const float* bk = (const float*)d_in[7];
    const float* Wv   = (const float*)d_in[8];  const float* bv = (const float*)d_in[9];
    const float* gw1  = (const float*)d_in[10]; const float* gb1 = (const float*)d_in[11];
    const float* gw2  = (const float*)d_in[12]; const float* gb2 = (const float*)d_in[13];
    const float* tw1  = (const float*)d_in[14]; const float* tb1 = (const float*)d_in[15];
    const float* tw2  = (const float*)d_in[16]; const float* tb2 = (const float*)d_in[17];
    const float* gaw1 = (const float*)d_in[18]; const float* gab1 = (const float*)d_in[19];
    const float* gaw2 = (const float*)d_in[20]; const float* gab2 = (const float*)d_in[21];
    const float* Wo   = (const float*)d_in[22]; const float* bo = (const float*)d_in[23];
    const float* lng  = (const float*)d_in[24]; const float* lnb = (const float*)d_in[25];

    // ---- workspace layout: f32 region then f16 region ----
    float* ws    = (float*)d_ws;
    float* gateh = ws;                          // 1024*1024
    float* gate  = gateh + NN * NN;             // 1024*2
    float* adj   = gate  + NN * 2;              // 1024*1024
    float* Pa    = adj   + NN * NN;             // 1024*1024
    float* Pb    = Pa    + NN * NN;             // 1024*1024
    float* abias = Pb    + NN * NN;             // 1024*8*1024
    float* out2  = abias + (size_t)NN * NH * NN;   // 1024*256

    _Float16* hbase = (_Float16*)(out2 + NN * HID);
    _Float16* srcH  = hbase;                    // 1024*256
    _Float16* tgtH  = srcH  + NN * HID;
    _Float16* catH  = tgtH  + NN * HID;         // 1024*512
    _Float16* adjH  = catH  + NN * 2 * HID;     // 1024*1024 (row-major A)
    _Float16* adjT  = adjH  + NN * NN;          // 1024*1024 (transposed B)
    _Float16* qH    = adjT  + NN * NN;          // 1024*256
    _Float16* kH    = qH    + NN * HID;
    _Float16* vT    = kH    + NN * HID;         // 256*1024 transposed V
    _Float16* PaH   = vT    + NN * HID;         // 1024*1024
    _Float16* PbH   = PaH   + NN * NN;
    _Float16* WqT   = PbH   + NN * NN;          // 256*256
    _Float16* WkT   = WqT   + HID * HID;
    _Float16* WvT   = WkT   + HID * HID;
    _Float16* WoT   = WvT   + HID * HID;
    _Float16* g1T   = WoT   + HID * HID;        // 1024*512 (N x K)
    _Float16* aoutH = g1T   + 4 * HID * 2 * HID;   // 1024*256

    // adjacency
    zero_kernel<<<1024, 256, 0, stream>>>(adj, NN * NN);
    scatter_kernel<<<NE / 256, 256, 0, stream>>>(ei, adj);

    // f16 pre-conversion (A operands row-major, B operands transposed)
    cvt_h_kernel<<<NN * HID / 256, 256, 0, stream>>>(src, srcH, NN * HID);
    cvt_h_kernel<<<NN * HID / 256, 256, 0, stream>>>(tgt, tgtH, NN * HID);
    cvt_h_kernel<<<NN * NN / 256, 256, 0, stream>>>(adj, adjH, NN * NN);
    cat_h_kernel<<<NN * 2 * HID / 256, 256, 0, stream>>>(src, tgt, catH);
    transpose_h_kernel<<<HID * HID / 256, 256, 0, stream>>>(Wq, WqT, HID, HID);
    transpose_h_kernel<<<HID * HID / 256, 256, 0, stream>>>(Wk, WkT, HID, HID);
    transpose_h_kernel<<<HID * HID / 256, 256, 0, stream>>>(Wv, WvT, HID, HID);
    transpose_h_kernel<<<HID * HID / 256, 256, 0, stream>>>(Wo, WoT, HID, HID);
    transpose_h_kernel<<<4 * HID * 2 * HID / 256, 256, 0, stream>>>(gaw1, g1T, 2 * HID, 4 * HID);
    transpose_h_kernel<<<NN * NN / 256, 256, 0, stream>>>(adj, adjT, NN, NN);

    // projections (f16 WMMA, double-buffered async-LDS staged B)
    gemm_kernel<0, false, 1><<<32, 256, 0, stream>>>(srcH, WqT, bq, nullptr, nullptr, qH, NN, HID, HID);
    gemm_kernel<0, false, 1><<<32, 256, 0, stream>>>(tgtH, WkT, bk, nullptr, nullptr, kH, NN, HID, HID);
    gemm_kernel<0, false, 2><<<32, 256, 0, stream>>>(tgtH, WvT, bv, nullptr, nullptr, vT, NN, HID, HID);

    // gate MLP
    gemm_kernel<1, true, 0><<<128, 256, 0, stream>>>(catH, g1T, gab1, nullptr, gateh, nullptr, NN, 4 * HID, 2 * HID);
    gate2_kernel<<<NN / 256, 256, 0, stream>>>(gateh, gaw2, gab2, gate);

    // path propagation: 3 x (P = min(P, P @ adj)); 0/1 data -> f16 exact
    gemm_kernel<2, true, 1><<<128, 256, 0, stream>>>(adjH, adjT, nullptr, adj, Pa, PaH, NN, NN, NN);
    gemm_kernel<2, true, 1><<<128, 256, 0, stream>>>(PaH,  adjT, nullptr, Pa,  Pb, PbH, NN, NN, NN);
    gemm_kernel<2, true, 1><<<128, 256, 0, stream>>>(PbH,  adjT, nullptr, Pb,  Pa, PaH, NN, NN, NN);

    // per-pair geo/topo MLPs + gate mix
    biasfuse_kernel<<<NN * NN / 256, 256, 0, stream>>>(pos, Pa, gate,
                                                       gw1, gb1, gw2, gb2,
                                                       tw1, tb1, tw2, tb2, abias);

    // fused flash attention (WMMA scores + WMMA P@V) -> f16
    attn_kernel<<<NN / 16, 256, 0, stream>>>(qH, kH, vT, abias, aoutH);

    // output projection + residual + LayerNorm
    gemm_kernel<0, true, 0><<<32, 256, 0, stream>>>(aoutH, WoT, bo, nullptr, out2, nullptr, NN, HID, HID);
    resid_ln_kernel<<<NN / 8, 256, 0, stream>>>(src, out2, lng, lnb, (float*)d_out);
}
